// HistorySAGE_39522289058164
// MI455X (gfx1250) — compile-verified, matrix operand order checked by reading, not run on hardware
//
#include <hip/hip_runtime.h>
#include <hip/hip_bf16.h>
#include <math.h>

// ---------------------------------------------------------------------------
// GraphSAGE 3-layer inference for MI455X (gfx1250, wave32, WMMA).
//
// Strategy (bandwidth-bound: ~27 GFLOP vs ~300MB traffic @ 23.3 TB/s):
//  * Fuse  agg@W + self@R  into one GEMM:  [agg ‖ self] @ [W;R], K = 2F.
//  * bf16 operands / f32 accumulation via v_wmma_f32_16x16x32_bf16.
//  * Aggregation + weight-prep kernels emit data PRE-SWIZZLED into the exact
//    CDNA5 WMMA 16-bit operand VGPR layout (ISA 7.12.2): GEMM inner loop is
//    pure contiguous 32B/lane b128 loads + WMMA.
//  * Each wave produces a 32(M)x64(N) tile: the 4 B fragments are reused by
//    two A fragments -> 8 WMMAs per 6 b128-load-pairs per K-step (double the
//    arithmetic intensity of a 16x64 tile).
// ---------------------------------------------------------------------------

typedef __attribute__((ext_vector_type(16))) __bf16 v16bf;
typedef __attribute__((ext_vector_type(8)))  float  v8f;

#define DEG 10
#define BN_EPS 1e-5f

// Static shapes from the reference setup (graph capture cannot read the
// nn*/ptr device scalars; these match setup_inputs exactly).
#define N_IN   300000
#define F_IN   128
#define HID    256
#define NCLS   47
#define NCLSP  64          // padded to a multiple of 16 for WMMA tiles
#define NN2    100000
#define NN1    50000
#define NN1P   50016       // padded to a multiple of 32 for the 32xN GEMM tile
#define NN0    20000

// round-to-nearest-even float -> bf16 (bit pattern)
__device__ __forceinline__ unsigned short f2bf(float f) {
  unsigned int u = __float_as_uint(f);
  unsigned int r = u + 0x7FFFu + ((u >> 16) & 1u);
  return (unsigned short)(r >> 16);
}

// CDNA5 16-bit WMMA operand swizzle (A 16x32 and B 32x16 share the K-map;
// 'row16' is the M-row for A, the N-col for B):
//   lane = row16 + 16*g,  g = (ko>>3)&1,  e = (ko&7) | ((ko&16)>>1)
__device__ __forceinline__ void swz16(int row16, int ko, int& lane, int& e) {
  int g = (ko >> 3) & 1;
  e    = (ko & 7) | ((ko & 16) >> 1);
  lane = row16 + (g << 4);
}

// ---------------------------------------------------------------------------
// Weight prep: pack [W;R] (K=2F rows, N cols, zero-padded) into WMMA
// B-operand order; build per-channel bias / BN scale / BN shift.
// Packed B chunk index: ((kb*nTiles + nt)*32 + lane)*16 + e
// ---------------------------------------------------------------------------
__global__ void prep_layer_k(const float* __restrict__ W, const float* __restrict__ R,
                             const float* __restrict__ b,
                             const float* __restrict__ gma, const float* __restrict__ beta,
                             const float* __restrict__ rm,  const float* __restrict__ rv,
                             unsigned short* __restrict__ Bpk,
                             float* __restrict__ bias, float* __restrict__ scale,
                             float* __restrict__ shift,
                             int F, int N, int Nvalid, int hasBN) {
  int K = 2 * F;
  int total = K * N;
  int nTiles = N >> 4;
  for (int t = blockIdx.x * blockDim.x + threadIdx.x; t < total;
       t += gridDim.x * blockDim.x) {
    int k = t / N, n = t % N;
    float v = 0.f;
    if (n < Nvalid)
      v = (k < F) ? W[(size_t)k * Nvalid + n] : R[(size_t)(k - F) * Nvalid + n];
    int kb = k >> 5, ko = k & 31, lane, e;
    swz16(n & 15, ko, lane, e);
    int nt = n >> 4;
    Bpk[(((size_t)kb * nTiles + nt) * 32 + lane) * 16 + e] = f2bf(v);
    if (t < N) {  // k==0, n==t : one thread per output channel
      bias[t] = (t < Nvalid) ? b[t] : 0.f;
      if (hasBN) {
        float sc = gma[t] * rsqrtf(rv[t] + BN_EPS);
        scale[t] = sc;
        shift[t] = beta[t] - rm[t] * sc;
      } else {
        scale[t] = 1.f;
        shift[t] = 0.f;
      }
    }
  }
}

// ---------------------------------------------------------------------------
// Segment-mean (uniform degree 10) fused with self-concat + bf16 conversion,
// written in WMMA A-operand order. Rows >= 'rows' (M-padding) are zero-filled.
// Packed A chunk index: ((mt*kBlocks + kb)*32 + lane)*16 + e
// ---------------------------------------------------------------------------
__global__ void agg_concat_k(const float* __restrict__ X, const int* __restrict__ idx,
                             unsigned short* __restrict__ Apk, int F, int kBlocks,
                             int rows) {
  int m = blockIdx.x;
  int valid = (m < rows);
  __shared__ int nb[DEG];
  if (valid && threadIdx.x < DEG) nb[threadIdx.x] = idx[m * DEG + threadIdx.x];
  __syncthreads();
  int K = F * 2;
  int mt = m >> 4, r = m & 15;
  for (int j = threadIdx.x; j < K; j += blockDim.x) {
    float v = 0.f;
    if (valid) {
      if (j < F) {
        float s = 0.f;
#pragma unroll
        for (int d = 0; d < DEG; ++d) s += X[(size_t)nb[d] * F + j];
        v = s * (1.0f / DEG);
      } else {
        v = X[(size_t)m * F + (j - F)];
      }
    }
    int kb = j >> 5, ko = j & 31, lane, e;
    swz16(r, ko, lane, e);
    Apk[(((size_t)mt * kBlocks + kb) * 32 + lane) * 16 + e] = f2bf(v);
  }
}

// ---------------------------------------------------------------------------
// WMMA GEMM: C = Apk @ Bpk, epilogue h = (dot + bias)*scale + shift,
// optional ReLU, fp32 row-major output (ldOut = N).
// One wave -> 32(M)x64(N) tile: 2 A fragments x 4 B fragments = 8 WMMAs/K-step.
// mPairs = (#16-row M tiles)/2 ; M padded so this is exact.
// ---------------------------------------------------------------------------
__global__ void gemm_wmma_k(const unsigned short* __restrict__ Apk,
                            const unsigned short* __restrict__ Bpk,
                            float* __restrict__ out,
                            int mPairs, int kBlocks, int nTiles,
                            const float* __restrict__ bias,
                            const float* __restrict__ scale,
                            const float* __restrict__ shift,
                            int relu, int ldOut) {
  const int lane = threadIdx.x & 31;
  const int wave = threadIdx.x >> 5;
  const int n64Tiles = nTiles >> 2;
  const int gw = blockIdx.x * (blockDim.x >> 5) + wave;
  if (gw >= mPairs * n64Tiles) return;  // wave-uniform: EXEC all-1s for WMMA
  const int tm = gw / n64Tiles;         // pair of 16-row M tiles
  const int tn64 = gw % n64Tiles;

  v8f acc00 = {}, acc01 = {}, acc02 = {}, acc03 = {};
  v8f acc10 = {}, acc11 = {}, acc12 = {}, acc13 = {};
  const unsigned short* aBase = Apk + (size_t)(2 * tm) * kBlocks * 512;  // 32 lanes*16
  const size_t aSecond = (size_t)kBlocks * 512;                          // next M tile

  for (int kb = 0; kb < kBlocks; ++kb) {
    if (kb + 2 < kBlocks)  // gfx1250 global_prefetch_b8 on the A stream
      __builtin_prefetch(aBase + (size_t)(kb + 2) * 512 + lane * 16, 0, 1);
    v16bf a0 = *(const v16bf*)(aBase + (size_t)kb * 512 + lane * 16);
    v16bf a1 = *(const v16bf*)(aBase + aSecond + (size_t)kb * 512 + lane * 16);
    size_t bo = (((size_t)kb * nTiles + tn64 * 4) * 32 + lane) * 16;
    v16bf b0 = *(const v16bf*)(Bpk + bo);
    v16bf b1 = *(const v16bf*)(Bpk + bo + 512);
    v16bf b2 = *(const v16bf*)(Bpk + bo + 1024);
    v16bf b3 = *(const v16bf*)(Bpk + bo + 1536);
    acc00 = __builtin_amdgcn_wmma_f32_16x16x32_bf16(false, a0, false, b0, (short)0, acc00, false, false);
    acc01 = __builtin_amdgcn_wmma_f32_16x16x32_bf16(false, a0, false, b1, (short)0, acc01, false, false);
    acc02 = __builtin_amdgcn_wmma_f32_16x16x32_bf16(false, a0, false, b2, (short)0, acc02, false, false);
    acc03 = __builtin_amdgcn_wmma_f32_16x16x32_bf16(false, a0, false, b3, (short)0, acc03, false, false);
    acc10 = __builtin_amdgcn_wmma_f32_16x16x32_bf16(false, a1, false, b0, (short)0, acc10, false, false);
    acc11 = __builtin_amdgcn_wmma_f32_16x16x32_bf16(false, a1, false, b1, (short)0, acc11, false, false);
    acc12 = __builtin_amdgcn_wmma_f32_16x16x32_bf16(false, a1, false, b2, (short)0, acc12, false, false);
    acc13 = __builtin_amdgcn_wmma_f32_16x16x32_bf16(false, a1, false, b3, (short)0, acc13, false, false);
  }

  // C/D layout: lane = N + 16*(M>>3), vgpr v = M&7
  const int nc = lane & 15;
  const int mrow = (lane >> 4) * 8;
  v8f accs[2][4] = {{acc00, acc01, acc02, acc03}, {acc10, acc11, acc12, acc13}};
#pragma unroll
  for (int t = 0; t < 4; ++t) {
    int n = tn64 * 64 + t * 16 + nc;
    float bs = bias[n], sc = scale[n], sh = shift[n];
#pragma unroll
    for (int half = 0; half < 2; ++half) {
      int rowBase = tm * 32 + half * 16 + mrow;
#pragma unroll
      for (int v = 0; v < 8; ++v) {
        float h = accs[half][t][v] + bs;
        h = h * sc + sh;
        if (relu) h = fmaxf(h, 0.f);
        out[(size_t)(rowBase + v) * ldOut + n] = h;
      }
    }
  }
}

// ---------------------------------------------------------------------------
// log_softmax over 47 classes, one wave (32 lanes) per row.
// ---------------------------------------------------------------------------
__global__ void logsoftmax_k(const float* __restrict__ logits, float* __restrict__ out,
                             int rows, int ldIn, int C) {
  int row = blockIdx.x * (blockDim.x >> 5) + (threadIdx.x >> 5);
  int lane = threadIdx.x & 31;
  if (row >= rows) return;
  const float* p = logits + (size_t)row * ldIn;
  float v0 = (lane < C) ? p[lane] : -INFINITY;
  float v1 = (lane + 32 < C) ? p[lane + 32] : -INFINITY;
  float m = fmaxf(v0, v1);
#pragma unroll
  for (int off = 16; off; off >>= 1) m = fmaxf(m, __shfl_xor(m, off, 32));
  float s = 0.f;
  if (lane < C) s += expf(v0 - m);
  if (lane + 32 < C) s += expf(v1 - m);
#pragma unroll
  for (int off = 16; off; off >>= 1) s += __shfl_xor(s, off, 32);
  float ls = logf(s);
  if (lane < C) out[(size_t)row * C + lane] = (v0 - m) - ls;
  if (lane + 32 < C) out[(size_t)row * C + lane + 32] = (v1 - m) - ls;
}

// ---------------------------------------------------------------------------
// Host-side orchestration (graph-capture safe: kernel launches only).
// ---------------------------------------------------------------------------
extern "C" void kernel_launch(void* const* d_in, const int* in_sizes, int n_in,
                              void* d_out, int out_size, void* d_ws, size_t ws_size,
                              hipStream_t stream) {
  (void)in_sizes; (void)n_in; (void)out_size; (void)ws_size;
  const float* x   = (const float*)d_in[0];
  const int*   idx = (const int*)d_in[2];
  const float* W0  = (const float*)d_in[3];
  const float* b0  = (const float*)d_in[4];
  const float* R0  = (const float*)d_in[5];
  const float* W1  = (const float*)d_in[6];
  const float* b1  = (const float*)d_in[7];
  const float* R1  = (const float*)d_in[8];
  const float* W2  = (const float*)d_in[9];
  const float* b2  = (const float*)d_in[10];
  const float* R2  = (const float*)d_in[11];
  const float* g0  = (const float*)d_in[12];
  const float* be0 = (const float*)d_in[13];
  const float* rm0 = (const float*)d_in[14];
  const float* rv0 = (const float*)d_in[15];
  const float* g1  = (const float*)d_in[16];
  const float* be1 = (const float*)d_in[17];
  const float* rm1 = (const float*)d_in[18];
  const float* rv1 = (const float*)d_in[19];
  float* out = (float*)d_out;

  char* ws = (char*)d_ws;
  // ---- workspace layout (bytes, 256-aligned) ----
  const size_t OFF_B0 = 0;                               // 256*256*2 = 131072
  const size_t OFF_B1 = OFF_B0 + 131072;                 // 512*256*2 = 262144
  const size_t OFF_B2 = OFF_B1 + 262144;                 // 512*64*2  = 65536
  size_t po = OFF_B2 + 65536;
  const size_t OFF_BI0 = po;        po += 1024;
  const size_t OFF_SC0 = po;        po += 1024;
  const size_t OFF_SH0 = po;        po += 1024;
  const size_t OFF_BI1 = po;        po += 1024;
  const size_t OFF_SC1 = po;        po += 1024;
  const size_t OFF_SH1 = po;        po += 1024;
  const size_t OFF_BI2 = po;        po += 256;
  const size_t OFF_SC2 = po;        po += 256;
  const size_t OFF_SH2 = po;        po += 256;
  // bf16 concat A, reused per layer: max(100000*256, 50016*512, 20000*512)*2
  const size_t OFF_A  = 524288;                          // 51,216,384 B
  const size_t OFF_H0 = OFF_A + 51216384;                // fp32 h0: 100000*256*4
  const size_t OFF_H1 = OFF_H0 + 102400000;              // fp32 h1: 50016*256*4
  const size_t OFF_LG = OFF_H0;                          // logits reuse h0 region
  // total ~205.4 MB

  unsigned short* B0p = (unsigned short*)(ws + OFF_B0);
  unsigned short* B1p = (unsigned short*)(ws + OFF_B1);
  unsigned short* B2p = (unsigned short*)(ws + OFF_B2);
  unsigned short* Ap  = (unsigned short*)(ws + OFF_A);
  float* h0 = (float*)(ws + OFF_H0);
  float* h1 = (float*)(ws + OFF_H1);
  float* lg = (float*)(ws + OFF_LG);

  // --- pack weights + per-channel epilogue params ---
  prep_layer_k<<<256, 256, 0, stream>>>(W0, R0, b0, g0, be0, rm0, rv0, B0p,
      (float*)(ws + OFF_BI0), (float*)(ws + OFF_SC0), (float*)(ws + OFF_SH0),
      F_IN, HID, HID, 1);
  prep_layer_k<<<512, 256, 0, stream>>>(W1, R1, b1, g1, be1, rm1, rv1, B1p,
      (float*)(ws + OFF_BI1), (float*)(ws + OFF_SC1), (float*)(ws + OFF_SH1),
      HID, HID, HID, 1);
  prep_layer_k<<<128, 256, 0, stream>>>(W2, R2, b2, nullptr, nullptr, nullptr, nullptr, B2p,
      (float*)(ws + OFF_BI2), (float*)(ws + OFF_SC2), (float*)(ws + OFF_SH2),
      HID, NCLSP, NCLS, 0);

  // --- layer 0: [100000, 2*128] -> [100000, 256], BN+ReLU ---
  agg_concat_k<<<NN2, 256, 0, stream>>>(x, idx, Ap, F_IN, (2 * F_IN) / 32, NN2);
  {
    int mPairs = NN2 / 32, kBlocks = (2 * F_IN) / 32, nTiles = HID / 16;
    int waves = mPairs * (nTiles / 4);
    gemm_wmma_k<<<(waves + 7) / 8, 256, 0, stream>>>(Ap, B0p, h0,
        mPairs, kBlocks, nTiles,
        (float*)(ws + OFF_BI0), (float*)(ws + OFF_SC0), (float*)(ws + OFF_SH0),
        1, HID);
  }

  // --- layer 1: [50000(->50016), 2*256] -> [50016, 256], BN+ReLU ---
  agg_concat_k<<<NN1P, 256, 0, stream>>>(h0, idx, Ap, HID, (2 * HID) / 32, NN1);
  {
    int mPairs = NN1P / 32, kBlocks = (2 * HID) / 32, nTiles = HID / 16;
    int waves = mPairs * (nTiles / 4);
    gemm_wmma_k<<<(waves + 7) / 8, 256, 0, stream>>>(Ap, B1p, h1,
        mPairs, kBlocks, nTiles,
        (float*)(ws + OFF_BI1), (float*)(ws + OFF_SC1), (float*)(ws + OFF_SH1),
        1, HID);
  }

  // --- layer 2: [20000, 2*256] -> [20000, 64(pad of 47)] ---
  agg_concat_k<<<NN0, 256, 0, stream>>>(h1, idx, Ap, HID, (2 * HID) / 32, NN0);
  {
    int mPairs = NN0 / 32, kBlocks = (2 * HID) / 32, nTiles = NCLSP / 16;
    int waves = mPairs * (nTiles / 4);
    gemm_wmma_k<<<(waves + 7) / 8, 256, 0, stream>>>(Ap, B2p, lg,
        mPairs, kBlocks, nTiles,
        (float*)(ws + OFF_BI2), (float*)(ws + OFF_SC2), (float*)(ws + OFF_SH2),
        0, NCLSP);
  }

  // --- log_softmax over 47 classes ---
  logsoftmax_k<<<(NN0 + 7) / 8, 256, 0, stream>>>(lg, out, NN0, NCLSP, NCLS);
}
